// QAttention_without_softmax_42769284334290
// MI455X (gfx1250) — 8-bit matrix operand order repaired and checked by reading, compiled, not run on hardware
//
#include <hip/hip_runtime.h>
#include <stdint.h>


typedef float v8f __attribute__((ext_vector_type(8)));
typedef float v2f __attribute__((ext_vector_type(2)));
typedef int   v8i __attribute__((ext_vector_type(8)));

#define BB 4
#define NN 2048
#define CC 768
#define HH 12
#define DD 64
#define QKV_ELEMS (BB*HH*NN*DD)
#define BH_STRIDE (NN*DD)

__global__ __launch_bounds__(256) void qkv_gemm_quant(
    const float* __restrict__ x, const float* __restrict__ W,
    const float* __restrict__ ps_q, const float* __restrict__ ps_k, const float* __restrict__ ps_v,
    int8_t* __restrict__ qg, int8_t* __restrict__ kg, int8_t* __restrict__ vtg)
{
  constexpr int AS_LD = 132, BS_LD = 68;
  __shared__ float As[32 * AS_LD];
  __shared__ float Bs[32 * BS_LD];

  const int tid  = threadIdx.x;
  const int lane = tid & 31;
  const int wave = tid >> 5;
  const int mBase = blockIdx.y * 128;
  const int nBase = blockIdx.x * 64;
  const int m0w = (wave >> 1) * 32;
  const int n0w = (wave & 1) * 32;
  const int rowc = lane & 15;
  const int hi   = lane >> 4;

  const float sq = ps_q[0], skk = ps_k[0], sv = ps_v[0];

  v8f acc[2][2] = {};

  for (int kc = 0; kc < 768; kc += 32) {
    for (int i = 0; i < 4; ++i) {
      int li = tid + i * 256;
      int r = li >> 3, kq = (li & 7) * 4;
      const float4 g = *(const float4*)(x + (size_t)(mBase + r) * 768 + kc + kq);
      if (kc + 32 < 768)
        __builtin_prefetch(x + (size_t)(mBase + r) * 768 + kc + 32 + kq, 0, 2);
      As[(kq + 0) * AS_LD + r] = g.x; As[(kq + 1) * AS_LD + r] = g.y;
      As[(kq + 2) * AS_LD + r] = g.z; As[(kq + 3) * AS_LD + r] = g.w;
    }
    for (int i = 0; i < 2; ++i) {
      int li = tid + i * 256;
      int nn2 = li >> 3, kq = (li & 7) * 4;
      const float4 g = *(const float4*)(W + (size_t)(nBase + nn2) * 768 + kc + kq);
      if (kc + 32 < 768)
        __builtin_prefetch(W + (size_t)(nBase + nn2) * 768 + kc + 32 + kq, 0, 2);
      Bs[(kq + 0) * BS_LD + nn2] = g.x; Bs[(kq + 1) * BS_LD + nn2] = g.y;
      Bs[(kq + 2) * BS_LD + nn2] = g.z; Bs[(kq + 3) * BS_LD + nn2] = g.w;
    }
    __syncthreads();
    for (int k4 = 0; k4 < 32; k4 += 4) {
      const int ks0 = k4 + 2 * hi, ks1 = ks0 + 1;
      v2f a[2], b[2];
      a[0].x = As[ks0 * AS_LD + m0w + rowc];      a[0].y = As[ks1 * AS_LD + m0w + rowc];
      a[1].x = As[ks0 * AS_LD + m0w + 16 + rowc]; a[1].y = As[ks1 * AS_LD + m0w + 16 + rowc];
      b[0].x = Bs[ks0 * BS_LD + n0w + rowc];      b[0].y = Bs[ks1 * BS_LD + n0w + rowc];
      b[1].x = Bs[ks0 * BS_LD + n0w + 16 + rowc]; b[1].y = Bs[ks1 * BS_LD + n0w + 16 + rowc];
      for (int tm = 0; tm < 2; ++tm)
        for (int tn = 0; tn < 2; ++tn)
          acc[tm][tn] = __builtin_amdgcn_wmma_f32_16x16x4_f32(
              false, a[tm], false, b[tn], (short)0, acc[tm][tn], false, false);
    }
    __syncthreads();
  }

  for (int tm = 0; tm < 2; ++tm)
    for (int tn = 0; tn < 2; ++tn)
      for (int i = 0; i < 8; ++i) {
        int gm = mBase + m0w + tm * 16 + i + 8 * hi;
        int gn = nBase + n0w + tn * 16 + rowc;
        float val = acc[tm][tn][i];
        int sec = gn / 768;
        int rem = gn - sec * 768;
        int h = rem >> 6, d = rem & 63;
        int b = gm >> 11, nr = gm & 2047;
        float s = (sec == 0) ? sq : (sec == 1 ? skk : sv);
        float qf = fminf(fmaxf(rintf(val / s), -4.f), 3.f);
        int8_t qv = (int8_t)(int)qf;
        size_t bh = (size_t)(b * HH + h);
        if (sec == 2)      vtg[bh * BH_STRIDE + (size_t)d * NN + nr] = qv;
        else if (sec == 1) kg [bh * BH_STRIDE + (size_t)nr * DD + d] = qv;
        else               qg [bh * BH_STRIDE + (size_t)nr * DD + d] = qv;
      }
}

__global__ __launch_bounds__(256) void attn_iu8(
    const int8_t* __restrict__ qg, const int8_t* __restrict__ kg, const int8_t* __restrict__ vtg,
    const float* __restrict__ ps_q, const float* __restrict__ ps_k, const float* __restrict__ ps_v,
    const float* __restrict__ ps_attn, const float* __restrict__ ps_after,
    float* __restrict__ og)
{
  constexpr int ALD = 132;
  constexpr int KLD = 68;
  constexpr int VLD = 132;
  __shared__ uint8_t a_lds[8 * 16 * ALD];
  __shared__ uint8_t ks_lds[2][128 * KLD];
  __shared__ uint8_t vs_lds[2][64 * VLD];

  const int tid = threadIdx.x, lane = tid & 31, wave = tid >> 5;
  const int rowc = lane & 15, hi = lane >> 4;
  const int b = blockIdx.z, h = blockIdx.y;
  const size_t bh = (size_t)(b * HH + h);
  const int q0 = blockIdx.x * 128 + wave * 16;

  const float sq = ps_q[0], sk2 = ps_k[0], sv = ps_v[0];
  const float sattn = ps_attn[0], safter = ps_after[0];
  const float qscale = sq * sk2 * 0.125f / (2048.f * sattn);
  const float oscale = sattn * sv;
  const float inv_after = 1.f / safter;

  uint8_t* aw = a_lds + wave * 16 * ALD;

  auto issue_tiles = [&](int j0, int p) {
    const int8_t* kbase = kg + bh * BH_STRIDE + (size_t)j0 * DD;
    uint32_t ks_off = (uint32_t)(uintptr_t)&ks_lds[p][0];
    for (int i = 0; i < 2; ++i) {
      int t = tid + i * 256;
      uint32_t dst = ks_off + (uint32_t)((t >> 2) * KLD + (t & 3) * 16);
      const int8_t* g = kbase + t * 16;
      asm volatile("global_load_async_to_lds_b128 %0, %1, off"
                   :: "v"(dst), "v"(g) : "memory");
    }
    const int8_t* vbase = vtg + bh * BH_STRIDE + j0;
    uint32_t vs_off = (uint32_t)(uintptr_t)&vs_lds[p][0];
    for (int i = 0; i < 2; ++i) {
      int t = tid + i * 256;
      int d = t >> 3, c = (t & 7) * 16;
      uint32_t dst = vs_off + (uint32_t)(d * VLD + c);
      const int8_t* g = vbase + (size_t)d * NN + c;
      asm volatile("global_load_async_to_lds_b128 %0, %1, off"
                   :: "v"(dst), "v"(g) : "memory");
    }
  };

  v8i qa;
  {
    const int8_t* qrow = qg + bh * BH_STRIDE + (size_t)(q0 + rowc) * DD;
    for (int v = 0; v < 8; ++v) {
      int dbase = (v & 3) * 4 + (v >= 4 ? 32 : 0) + hi * 16;
      qa[v] = *(const int*)(qrow + dbase);
    }
  }

  v8i acc[4] = {};

  issue_tiles(0, 0);
  asm volatile("s_wait_asynccnt 0" ::: "memory");
  __syncthreads();

  int p = 0;
  for (int j0 = 0; j0 < NN; j0 += 128) {
    if (j0 + 128 < NN) issue_tiles(j0 + 128, p ^ 1);
    const uint8_t* ks = ks_lds[p];
    const uint8_t* vs = vs_lds[p];

    for (int tt = 0; tt < 8; ++tt) {
      v8i kf;
      const uint8_t* krow = ks + (size_t)(tt * 16 + rowc) * KLD;
      for (int v = 0; v < 8; ++v) {
        int dbase = (v & 3) * 4 + (v >= 4 ? 32 : 0) + hi * 16;
        kf[v] = *(const int*)(krow + dbase);
      }
      v8i S = {};
      S = __builtin_amdgcn_wmma_i32_16x16x64_iu8(true, qa, true, kf, S, false, false);
      for (int i = 0; i < 8; ++i) {
        int r = i + 8 * hi;
        int c = tt * 16 + rowc;
        int ai = (int)fminf(fmaxf(rintf((float)S[i] * qscale), 0.f), 7.f);
        aw[r * ALD + c] = (uint8_t)ai;
      }
    }

    v8i af0, af1;
    for (int v = 0; v < 8; ++v) {
      int kb = (v & 3) * 4 + (v >= 4 ? 32 : 0) + hi * 16;
      af0[v] = *(const int*)(aw + rowc * ALD + kb);
      af1[v] = *(const int*)(aw + rowc * ALD + kb + 64);
    }
    for (int dt = 0; dt < 4; ++dt) {
      const uint8_t* vcol = vs + (size_t)(dt * 16 + rowc) * VLD;
      v8i vf0, vf1;
      for (int v = 0; v < 8; ++v) {
        int kb = (v & 3) * 4 + (v >= 4 ? 32 : 0) + hi * 16;
        vf0[v] = *(const int*)(vcol + kb);
        vf1[v] = *(const int*)(vcol + kb + 64);
      }
      acc[dt] = __builtin_amdgcn_wmma_i32_16x16x64_iu8(false, af0, true, vf0, acc[dt], false, false);
      acc[dt] = __builtin_amdgcn_wmma_i32_16x16x64_iu8(false, af1, true, vf1, acc[dt], false, false);
    }

    asm volatile("s_wait_asynccnt 0" ::: "memory");
    __syncthreads();
    p ^= 1;
  }

  for (int dt = 0; dt < 4; ++dt)
    for (int i = 0; i < 8; ++i) {
      int qr = q0 + i + 8 * hi;
      int d = dt * 16 + rowc;
      float f = oscale * (float)acc[dt][i];
      float oq = fminf(fmaxf(rintf(f * inv_after), -4.f), 3.f);
      og[(size_t)(b * NN + qr) * CC + h * DD + d] = safter * oq;
    }
}

__global__ __launch_bounds__(256) void proj_gemm(
    const float* __restrict__ A, const float* __restrict__ W, const float* __restrict__ bias,
    float* __restrict__ out)
{
  constexpr int AS_LD = 132, BS_LD = 68;
  __shared__ float As[32 * AS_LD];
  __shared__ float Bs[32 * BS_LD];

  const int tid = threadIdx.x, lane = tid & 31, wave = tid >> 5;
  const int mBase = blockIdx.y * 128;
  const int nBase = blockIdx.x * 64;
  const int m0w = (wave >> 1) * 32;
  const int n0w = (wave & 1) * 32;
  const int rowc = lane & 15, hi = lane >> 4;

  v8f acc[2][2] = {};

  for (int kc = 0; kc < 768; kc += 32) {
    for (int i = 0; i < 4; ++i) {
      int li = tid + i * 256;
      int r = li >> 3, kq = (li & 7) * 4;
      const float4 g = *(const float4*)(A + (size_t)(mBase + r) * 768 + kc + kq);
      if (kc + 32 < 768)
        __builtin_prefetch(A + (size_t)(mBase + r) * 768 + kc + 32 + kq, 0, 2);
      As[(kq + 0) * AS_LD + r] = g.x; As[(kq + 1) * AS_LD + r] = g.y;
      As[(kq + 2) * AS_LD + r] = g.z; As[(kq + 3) * AS_LD + r] = g.w;
    }
    for (int i = 0; i < 2; ++i) {
      int li = tid + i * 256;
      int nn2 = li >> 3, kq = (li & 7) * 4;
      const float4 g = *(const float4*)(W + (size_t)(nBase + nn2) * 768 + kc + kq);
      if (kc + 32 < 768)
        __builtin_prefetch(W + (size_t)(nBase + nn2) * 768 + kc + 32 + kq, 0, 2);
      Bs[(kq + 0) * BS_LD + nn2] = g.x; Bs[(kq + 1) * BS_LD + nn2] = g.y;
      Bs[(kq + 2) * BS_LD + nn2] = g.z; Bs[(kq + 3) * BS_LD + nn2] = g.w;
    }
    __syncthreads();
    for (int k4 = 0; k4 < 32; k4 += 4) {
      const int ks0 = k4 + 2 * hi, ks1 = ks0 + 1;
      v2f a[2], b[2];
      a[0].x = As[ks0 * AS_LD + m0w + rowc];      a[0].y = As[ks1 * AS_LD + m0w + rowc];
      a[1].x = As[ks0 * AS_LD + m0w + 16 + rowc]; a[1].y = As[ks1 * AS_LD + m0w + 16 + rowc];
      b[0].x = Bs[ks0 * BS_LD + n0w + rowc];      b[0].y = Bs[ks1 * BS_LD + n0w + rowc];
      b[1].x = Bs[ks0 * BS_LD + n0w + 16 + rowc]; b[1].y = Bs[ks1 * BS_LD + n0w + 16 + rowc];
      for (int tm = 0; tm < 2; ++tm)
        for (int tn = 0; tn < 2; ++tn)
          acc[tm][tn] = __builtin_amdgcn_wmma_f32_16x16x4_f32(
              false, a[tm], false, b[tn], (short)0, acc[tm][tn], false, false);
    }
    __syncthreads();
  }

  for (int tm = 0; tm < 2; ++tm)
    for (int tn = 0; tn < 2; ++tn)
      for (int i = 0; i < 8; ++i) {
        int gm = mBase + m0w + tm * 16 + i + 8 * hi;
        int gn = nBase + n0w + tn * 16 + rowc;
        out[(size_t)gm * 768 + gn] = acc[tm][tn][i] + bias[gn];
      }
}

extern "C" void kernel_launch(void* const* d_in, const int* in_sizes, int n_in,
                              void* d_out, int out_size, void* d_ws, size_t ws_size,
                              hipStream_t stream)
{
  const float* x      = (const float*)d_in[0];
  const float* Wqkv   = (const float*)d_in[1];
  const float* Wproj  = (const float*)d_in[2];
  const float* bproj  = (const float*)d_in[3];
  const float* s_q    = (const float*)d_in[4];
  const float* s_k    = (const float*)d_in[5];
  const float* s_v    = (const float*)d_in[6];
  const float* s_attn = (const float*)d_in[7];
  const float* s_after= (const float*)d_in[8];
  float* out = (float*)d_out;

  int8_t* qg  = (int8_t*)d_ws;
  int8_t* kg  = qg + QKV_ELEMS;
  int8_t* vtg = kg + QKV_ELEMS;
  float*  og  = (float*)(vtg + QKV_ELEMS);

  qkv_gemm_quant<<<dim3(36, 64), 256, 0, stream>>>(x, Wqkv, s_q, s_k, s_v, qg, kg, vtg);
  attn_iu8<<<dim3(16, 12, 4), 256, 0, stream>>>(qg, kg, vtg, s_q, s_k, s_v, s_attn, s_after, og);
  proj_gemm<<<dim3(12, 64), 256, 0, stream>>>(og, Wproj, bproj, out);
}
